// EMM_19404662243509
// MI455X (gfx1250) — matrix operand order repaired and checked by reading, not compiled, not run on hardware
//
#include <hip/hip_runtime.h>

#define B_    8
#define S_    4096
#define D_    256
#define H_    256
#define FEAT_ 1794
#define KTOT  768

typedef __attribute__((ext_vector_type(16))) __bf16 v16bf;
typedef __attribute__((ext_vector_type(8)))  float  v8f;

union BFrag { uint4 u[2]; v16bf v; };
union AFrag { unsigned short s[16]; v16bf v; };

// f32 -> bf16 bits, round-to-nearest-even, pure integer domain
__device__ __forceinline__ unsigned short f2bf_bits(float x) {
  union { float f; unsigned int u; } cv; cv.f = x;
  unsigned int lsb = (cv.u >> 16) & 1u;
  unsigned int r = cv.u + 0x7fffu + lsb;
  return (unsigned short)(r >> 16);
}

// |d| -> bf16 bits: sign cleared in integer domain (avoids fabs/cvt DAG combine bug)
__device__ __forceinline__ unsigned short f2bf_abs_bits(float d) {
  union { float f; unsigned int u; } cv; cv.f = d;
  unsigned int a = cv.u & 0x7fffffffu;
  unsigned int lsb = (a >> 16) & 1u;
  unsigned int r = a + 0x7fffu + lsb;
  return (unsigned short)(r >> 16);
}

// ---------------------------------------------------------------------------
// One-time init: m0 = q ; awq[b,k] = sum_d attW[k,d] * q[b,d]
// ---------------------------------------------------------------------------
__global__ __launch_bounds__(256) void init_kernel(
    const float* __restrict__ q, const float* __restrict__ attW,
    float* __restrict__ mbuf, float* __restrict__ awq)
{
  __shared__ float sq[D_];
  const int b = blockIdx.x, t = threadIdx.x;
  float qv = q[b * D_ + t];
  sq[t] = qv;
  mbuf[b * D_ + t] = qv;
  __syncthreads();
  const float* ar = attW + (size_t)t * D_;
  float aw = 0.f;
  for (int d = 0; d < D_; ++d) aw += ar[d] * sq[d];
  awq[b * D_ + t] = aw;
}

// ---------------------------------------------------------------------------
// Per-iteration prep (per batch):
//   awm[b,k]   = sum_d attW[k,d]*m[b,d]
//   cvec[b,h]  = b1[h] + sum_d m[d]*w1[h,D+d] + sum_d q[d]*w1[h,2D+d]
//   BigWT[b,h,0:256]   = w1[h,k] + q[k]*w1[h,3D+k] + m[k]*w1[h,4D+k]   (bf16)
//   BigWT[b,h,256:512] = w1[h,5D+k]   ;   BigWT[b,h,512:768] = w1[h,6D+k]
// ---------------------------------------------------------------------------
__global__ __launch_bounds__(256) void prep_kernel(
    const float* __restrict__ q, const float* __restrict__ mbuf,
    const float* __restrict__ attW,
    const float* __restrict__ w1, const float* __restrict__ b1,
    float* __restrict__ awm, float* __restrict__ cvec,
    unsigned short* __restrict__ bigwt)
{
  __shared__ float sq[D_], sm[D_];
  const int b = blockIdx.x, t = threadIdx.x;
  sq[t] = q[b * D_ + t];
  sm[t] = mbuf[b * D_ + t];
  __syncthreads();

  const float* ar = attW + (size_t)t * D_;
  float aw = 0.f;
  for (int d = 0; d < D_; ++d) aw += ar[d] * sm[d];
  awm[b * D_ + t] = aw;

  const float* w1r = w1 + (size_t)t * FEAT_;
  float cv = b1[t];
  for (int d = 0; d < D_; ++d) cv += sm[d] * w1r[256 + d] + sq[d] * w1r[512 + d];
  cvec[b * H_ + t] = cv;

  unsigned short* dst = bigwt + ((size_t)b * H_ + t) * KTOT;
  for (int k = 0; k < 256; ++k)
    dst[k] = f2bf_bits(w1r[k] + sq[k] * w1r[768 + k] + sm[k] * w1r[1024 + k]);
  for (int k = 0; k < 256; ++k) dst[256 + k] = f2bf_bits(w1r[1280 + k]);
  for (int k = 0; k < 256; ++k) dst[512 + k] = f2bf_bits(w1r[1536 + k]);
}

// ---------------------------------------------------------------------------
// Main per-iteration kernel: scores[b,s] for all 32768 rows.
// Block = 128 rows of one batch, 8 waves; wave = 16 rows x 256 cols of h.
// K = 768 = [c | |c-q| | |c-m|] vs BigWT[b], bf16 WMMA f32-acc.
// Per-row dot_q/dot_m accumulated for free from the same c loads.
// ---------------------------------------------------------------------------
__global__ __launch_bounds__(256) void dmn_gemm_kernel(
    const float* __restrict__ c, const float* __restrict__ q,
    const float* __restrict__ w1, const float* __restrict__ w2,
    const float* __restrict__ b2,
    const unsigned short* __restrict__ bigwt,
    const float* __restrict__ cvec,
    const float* __restrict__ mbuf,
    const float* __restrict__ awq, const float* __restrict__ awm,
    float* __restrict__ scores)
{
  __shared__ __align__(16) unsigned short ldsBW[3][H_][32]; // 48 KB
  __shared__ float ldsQ[D_], ldsM[D_], ldsAq[D_], ldsAm[D_];

  const int t     = threadIdx.x;
  const int bb    = blockIdx.x >> 5;
  const int s0    = (blockIdx.x & 31) << 7;
  const int wave  = t >> 5;
  const int lane  = t & 31;
  const int l16   = lane & 15;
  const int hi    = lane >> 4;
  const int klo   = hi << 3;
  const int srow0 = s0 + (wave << 4);

  ldsQ[t]  = q[bb * D_ + t];
  ldsM[t]  = mbuf[bb * D_ + t];
  ldsAq[t] = awq[bb * D_ + t];
  ldsAm[t] = awm[bb * D_ + t];

  const float* crow = c + (size_t)(bb * S_ + srow0 + l16) * D_;
  const unsigned short* wsrc = bigwt + ((size_t)bb * H_ + t) * KTOT;

  v8f acc[16];
  #pragma unroll
  for (int i = 0; i < 16; ++i)
    #pragma unroll
    for (int r = 0; r < 8; ++r) acc[i][r] = 0.f;

  float pdq = 0.f, pdm = 0.f;

  for (int kk = 0; kk < 256; kk += 32) {
    __syncthreads();
    // cooperative stage of BigWT chunk (3 groups x 256 cols x 32 k) into LDS
    #pragma unroll
    for (int g = 0; g < 3; ++g) {
      const uint4* srcp = (const uint4*)(wsrc + g * 256 + kk);
      uint4* dstp = (uint4*)(&ldsBW[g][t][0]);
      dstp[0] = srcp[0]; dstp[1] = srcp[1]; dstp[2] = srcp[2]; dstp[3] = srcp[3];
    }
    __syncthreads();

    // load 16 c values for this lane's row (A-layout K split: klo..+8, 16+klo..+8)
    float cvv[16];
    *(float4*)(cvv + 0)  = *(const float4*)(crow + kk + klo);
    *(float4*)(cvv + 4)  = *(const float4*)(crow + kk + klo + 4);
    *(float4*)(cvv + 8)  = *(const float4*)(crow + kk + klo + 16);
    *(float4*)(cvv + 12) = *(const float4*)(crow + kk + klo + 20);

    AFrag fa, fb, fc;
    #pragma unroll
    for (int j = 0; j < 16; ++j) {
      int k = kk + klo + (j < 8 ? j : j + 8);
      float x = cvv[j];
      pdq += x * ldsAq[k];
      pdm += x * ldsAm[k];
      fa.s[j] = f2bf_bits(x);
      fb.s[j] = f2bf_abs_bits(x - ldsQ[k]);
      fc.s[j] = f2bf_abs_bits(x - ldsM[k]);
    }
    v16bf afr[3] = {fa.v, fb.v, fc.v};

    #pragma unroll
    for (int g = 0; g < 3; ++g) {
      #pragma unroll
      for (int nt = 0; nt < 16; ++nt) {
        BFrag bf;
        const uint4* p = (const uint4*)(&ldsBW[g][nt * 16 + l16][0]);
        bf.u[0] = p[hi]; bf.u[1] = p[2 + hi];
        acc[nt] = __builtin_amdgcn_wmma_f32_16x16x32_bf16(
            false, afr[g], false, bf.v, (short)0, acc[nt], false, false);
      }
    }
  }

  // finalize per-row dot_q/dot_m (lane L and L^16 each hold half the K range)
  float dq = pdq + __shfl_xor(pdq, 16, 32);
  float dm = pdm + __shfl_xor(pdm, 16, 32);
  float dqr[8], dmr[8];
  #pragma unroll
  for (int r = 0; r < 8; ++r) {
    int srcl = r + hi * 8;          // accumulator row M = r + hi*8
    dqr[r] = __shfl(dq, srcl, 32);
    dmr[r] = __shfl(dm, srcl, 32);
  }

  // epilogue: pre = acc + cvec[n] + dot_q*w1[n,7D] + dot_m*w1[n,7D+1]
  //           score_row += tanh(pre) * w2[n]
  float sp[8] = {0.f, 0.f, 0.f, 0.f, 0.f, 0.f, 0.f, 0.f};
  #pragma unroll
  for (int nt = 0; nt < 16; ++nt) {
    int n = nt * 16 + l16;
    float cvn  = cvec[bb * H_ + n];
    float wdqn = w1[(size_t)n * FEAT_ + 1792];
    float wdmn = w1[(size_t)n * FEAT_ + 1793];
    float w2n  = w2[n];
    #pragma unroll
    for (int r = 0; r < 8; ++r) {
      float pre = acc[nt][r] + cvn + dqr[r] * wdqn + dmr[r] * wdmn;
      sp[r] += tanhf(pre) * w2n;
    }
  }
  // reduce over the 16 lanes of each half-wave (masks stay within the half)
  #pragma unroll
  for (int off = 1; off < 16; off <<= 1)
    #pragma unroll
    for (int r = 0; r < 8; ++r) sp[r] += __shfl_xor(sp[r], off, 32);

  if (l16 == 0) {
    float bias2 = b2[0];
    #pragma unroll
    for (int r = 0; r < 8; ++r)
      scores[(size_t)bb * S_ + srow0 + hi * 8 + r] = sp[r] + bias2;
  }
}

// ---------------------------------------------------------------------------
// Masked softmax over S per batch, write att to d_out, compute e = sum att*c
// ---------------------------------------------------------------------------
__global__ __launch_bounds__(256) void softmax_e_kernel(
    const float* __restrict__ scores, const float* __restrict__ c,
    const int* __restrict__ len_c,
    float* __restrict__ att_out, float* __restrict__ ebuf)
{
  __shared__ float red[256];
  __shared__ float attl[256];
  const int b = blockIdx.x, t = threadIdx.x;
  const int len = len_c[b];
  const float* sc = scores + (size_t)b * S_;

  float mx = -3.4e38f;
  for (int s = t; s < len; s += 256) mx = fmaxf(mx, sc[s]);
  red[t] = mx; __syncthreads();
  for (int o = 128; o > 0; o >>= 1) {
    if (t < o) red[t] = fmaxf(red[t], red[t + o]);
    __syncthreads();
  }
  mx = red[0]; __syncthreads();

  float sm = 0.f;
  for (int s = t; s < len; s += 256) sm += expf(sc[s] - mx);
  red[t] = sm; __syncthreads();
  for (int o = 128; o > 0; o >>= 1) {
    if (t < o) red[t] += red[t + o];
    __syncthreads();
  }
  float inv = 1.f / red[0]; __syncthreads();

  float ev = 0.f;   // e[b, d=t]
  for (int s0 = 0; s0 < S_; s0 += 256) {
    int s = s0 + t;
    float a = (s < len) ? expf(sc[s] - mx) * inv : 0.f;
    attl[t] = a;
    att_out[(size_t)b * S_ + s] = a;
    __syncthreads();
    if (s0 < len) {
      const float* cbase = c + ((size_t)b * S_ + s0) * D_ + t;
      for (int j = 0; j < 256; ++j) ev += attl[j] * cbase[(size_t)j * D_];
    }
    __syncthreads();
  }
  ebuf[b * D_ + t] = ev;
}

// ---------------------------------------------------------------------------
// GRU cell: m = gru(e, m)
// ---------------------------------------------------------------------------
__global__ __launch_bounds__(256) void gru_kernel(
    const float* __restrict__ ebuf, float* __restrict__ mbuf,
    const float* __restrict__ w_ih, const float* __restrict__ w_hh,
    const float* __restrict__ b_ih, const float* __restrict__ b_hh,
    float* __restrict__ out_m, int write_out)
{
  __shared__ float se[D_], sm[D_];
  const int b = blockIdx.x, d = threadIdx.x;
  se[d] = ebuf[b * D_ + d];
  sm[d] = mbuf[b * D_ + d];
  __syncthreads();

  const float* wr = w_ih + (size_t)d * D_;
  const float* wz = w_ih + (size_t)(D_ + d) * D_;
  const float* wn = w_ih + (size_t)(2 * D_ + d) * D_;
  const float* vr = w_hh + (size_t)d * D_;
  const float* vz = w_hh + (size_t)(D_ + d) * D_;
  const float* vn = w_hh + (size_t)(2 * D_ + d) * D_;
  float gir = b_ih[d], giz = b_ih[D_ + d], gin = b_ih[2 * D_ + d];
  float ghr = b_hh[d], ghz = b_hh[D_ + d], ghn = b_hh[2 * D_ + d];
  for (int k = 0; k < D_; ++k) {
    float ek = se[k], mk = sm[k];
    gir += ek * wr[k]; giz += ek * wz[k]; gin += ek * wn[k];
    ghr += mk * vr[k]; ghz += mk * vz[k]; ghn += mk * vn[k];
  }
  float r = 1.f / (1.f + expf(-(gir + ghr)));
  float z = 1.f / (1.f + expf(-(giz + ghz)));
  float n = tanhf(gin + r * ghn);
  float mn = (1.f - z) * n + z * sm[d];
  __syncthreads();
  mbuf[b * D_ + d] = mn;
  if (write_out) out_m[b * D_ + d] = mn;
}

// ---------------------------------------------------------------------------
extern "C" void kernel_launch(void* const* d_in, const int* in_sizes, int n_in,
                              void* d_out, int out_size, void* d_ws, size_t ws_size,
                              hipStream_t stream)
{
  const float* c    = (const float*)d_in[0];
  const float* q    = (const float*)d_in[1];
  const int*   lenc = (const int*)  d_in[2];
  const float* attW = (const float*)d_in[3];
  const float* w1   = (const float*)d_in[4];
  const float* b1   = (const float*)d_in[5];
  const float* w2   = (const float*)d_in[6];
  const float* b2   = (const float*)d_in[7];
  const float* w_ih = (const float*)d_in[8];
  const float* w_hh = (const float*)d_in[9];
  const float* b_ih = (const float*)d_in[10];
  const float* b_hh = (const float*)d_in[11];
  (void)in_sizes; (void)n_in; (void)out_size; (void)ws_size;

  float* out = (float*)d_out;

  // workspace carve-out (all f32 except BigWT bf16): ~3.3 MB total
  float* ws     = (float*)d_ws;
  float* mbuf   = ws;                 // B*D
  float* awq    = ws + 2048;          // B*D
  float* awm    = ws + 4096;          // B*D
  float* cvec   = ws + 6144;          // B*H
  float* ebuf   = ws + 8192;          // B*D
  float* scores = ws + 10240;         // B*S
  unsigned short* bigwt = (unsigned short*)(ws + 10240 + B_ * S_); // B*H*768

  init_kernel<<<B_, 256, 0, stream>>>(q, attW, mbuf, awq);
  for (int it = 0; it < 3; ++it) {
    prep_kernel<<<B_, 256, 0, stream>>>(q, mbuf, attW, w1, b1, awm, cvec, bigwt);
    dmn_gemm_kernel<<<B_ * 32, 256, 0, stream>>>(c, q, w1, w2, b2, bigwt, cvec,
                                                 mbuf, awq, awm, scores);
    softmax_e_kernel<<<B_, 256, 0, stream>>>(scores, c, lenc,
                                             out + 2048 + (size_t)it * B_ * S_, ebuf);
    gru_kernel<<<B_, 256, 0, stream>>>(ebuf, mbuf, w_ih, w_hh, b_ih, b_hh,
                                       out, it == 2 ? 1 : 0);
  }
}
